// MiniGPT_75024488727234
// MI455X (gfx1250) — compile-verified, matrix operand order checked by reading, not run on hardware
//
#include <hip/hip_runtime.h>
#include <math.h>

typedef __attribute__((ext_vector_type(16))) __bf16 v16bf;
typedef __attribute__((ext_vector_type(8)))  __bf16 v8bf;
typedef __attribute__((ext_vector_type(8)))  float  v8f;

#define DI __device__ __forceinline__

static constexpr int Bz = 4, Tt = 1024, Ee = 1024, Ll = 8, Vv = 32000, Hh = 16, HD = 64;
static constexpr int Mrows = Bz * Tt;   // 4096
static constexpr int S3    = 3 * Ee;    // 3072

// ---------------------------------------------------------------------------
// WMMA helper: D = A(16x32 bf16) * B(32x16 bf16) + C(16x16 f32)
// ---------------------------------------------------------------------------
DI v8f wmma_bf16(v16bf a, v16bf b, v8f c) {
  return __builtin_amdgcn_wmma_f32_16x16x32_bf16(false, a, false, b, (short)0, c,
                                                 false, false);
}

// Load one operand fragment. Per CDNA5 16-bit operand layout:
// lane (r = lane&15, h = lane>>4) holds K = {8h..8h+7} in v0..3 and
// K = {16+8h..23+8h} in v4..7 -> two contiguous 16B chunks at row offsets
// off and off+16 (off = 8*h). Works for LDS or global pointers.
DI v16bf ld_frag(const __bf16* rowp, int off) {
  v8bf lo = *(const v8bf*)(rowp + off);
  v8bf hi = *(const v8bf*)(rowp + off + 16);
  v16bf r;
#pragma unroll
  for (int i = 0; i < 8; ++i) { r[i] = lo[i]; r[i + 8] = hi[i]; }
  return r;
}

// ---------------------------------------------------------------------------
// GEMM: out[M,N] = A[M,K](bf16) @ W[N,K](f32 -> bf16 in LDS) + bias
// Block tile 128x128, 8 waves (4x2), wave tile 32x64 (8 WMMA accums).
// Double-buffered LDS, one barrier per 32-K step; next tile's global loads
// are issued before the WMMA block so VMEM overlaps matrix math.
// MODE 0: bf16 out (+bias)     MODE 1: bf16 out, bias+GELU(exact)
// MODE 2: f32 out = resid + acc + bias      MODE 3: f32 out = acc
// ---------------------------------------------------------------------------
template <int MODE>
__global__ __launch_bounds__(256) void gemm_bf16_kernel(
    const __bf16* __restrict__ A, const float* __restrict__ W,
    const float* __restrict__ bias, const float* __restrict__ resid,
    __bf16* __restrict__ outb, float* __restrict__ outf, int N, int K) {
  __shared__ __align__(16) __bf16 sA[2][128 * 40];
  __shared__ __align__(16) __bf16 sB[2][128 * 40];

  const int tid  = threadIdx.x;
  const int lane = tid & 31;
  const int w    = tid >> 5;
  const int lm   = lane & 15, lh = lane >> 4;
  const int wm   = w >> 1, wn = w & 1;   // 4 x 2 wave grid
  const int m0   = blockIdx.y * 128;
  const int n0   = blockIdx.x * 128;

  v8f acc[2][4] = {};

  const int sr = tid >> 1;             // staged row 0..127
  const int sc = (tid & 1) * 16;       // staged col 0 / 16
  const __bf16* ga = A + (size_t)(m0 + sr) * K + sc;
  const float*  gw = W + (size_t)(n0 + sr) * K + sc;
  const int lofs = sr * 40 + sc;

  v8bf ra0, ra1, rb0, rb1;
  auto load_regs = [&](int k0) {
    ra0 = *(const v8bf*)(ga + k0);
    ra1 = *(const v8bf*)(ga + k0 + 8);
    float4 f0 = *(const float4*)(gw + k0);
    float4 f1 = *(const float4*)(gw + k0 + 4);
    float4 f2 = *(const float4*)(gw + k0 + 8);
    float4 f3 = *(const float4*)(gw + k0 + 12);
    rb0 = v8bf{(__bf16)f0.x, (__bf16)f0.y, (__bf16)f0.z, (__bf16)f0.w,
               (__bf16)f1.x, (__bf16)f1.y, (__bf16)f1.z, (__bf16)f1.w};
    rb1 = v8bf{(__bf16)f2.x, (__bf16)f2.y, (__bf16)f2.z, (__bf16)f2.w,
               (__bf16)f3.x, (__bf16)f3.y, (__bf16)f3.z, (__bf16)f3.w};
  };
  auto store_buf = [&](int buf) {
    *(v8bf*)(&sA[buf][lofs])     = ra0;
    *(v8bf*)(&sA[buf][lofs + 8]) = ra1;
    *(v8bf*)(&sB[buf][lofs])     = rb0;
    *(v8bf*)(&sB[buf][lofs + 8]) = rb1;
  };

  load_regs(0);
  store_buf(0);
  __syncthreads();

  const int nkt = K >> 5;
  for (int kt = 0; kt < nkt; ++kt) {
    const int cur = kt & 1;
    if (kt + 1 < nkt) load_regs((kt + 1) << 5);  // overlap with WMMA below

    v16bf af[2], bfv[4];
#pragma unroll
    for (int mt = 0; mt < 2; ++mt)
      af[mt] = ld_frag(&sA[cur][(wm * 32 + mt * 16 + lm) * 40], lh * 8);
#pragma unroll
    for (int nt = 0; nt < 4; ++nt)
      bfv[nt] = ld_frag(&sB[cur][(wn * 64 + nt * 16 + lm) * 40], lh * 8);
#pragma unroll
    for (int mt = 0; mt < 2; ++mt)
#pragma unroll
      for (int nt = 0; nt < 4; ++nt)
        acc[mt][nt] = wmma_bf16(af[mt], bfv[nt], acc[mt][nt]);

    if (kt + 1 < nkt) {
      store_buf(cur ^ 1);   // safe: buf cur^1 was last read before prev barrier
      __syncthreads();
    }
  }

  // Epilogue. C layout: element (m,n): vgpr r = m&7, lane = (n&15) + 16*(m>>3)
#pragma unroll
  for (int mt = 0; mt < 2; ++mt) {
#pragma unroll
    for (int nt = 0; nt < 4; ++nt) {
      const int n  = n0 + wn * 64 + nt * 16 + lm;
      const float bv = (MODE == 3) ? 0.0f : bias[n];
#pragma unroll
      for (int r = 0; r < 8; ++r) {
        const int m = m0 + wm * 32 + mt * 16 + r + 8 * lh;
        float v = acc[mt][nt][r] + bv;
        const size_t off = (size_t)m * N + n;
        if constexpr (MODE == 0) {
          outb[off] = (__bf16)v;
        } else if constexpr (MODE == 1) {
          v = 0.5f * v * (1.0f + erff(v * 0.70710678118654752f));
          outb[off] = (__bf16)v;
        } else if constexpr (MODE == 2) {
          outf[off] = resid[off] + v;
        } else {
          outf[off] = v;
        }
      }
    }
  }
}

// ---------------------------------------------------------------------------
// Flash attention: one block per (b, h, 64-row q tile); 4 waves x 16 q rows.
// K block staged row-major in LDS (shared by all waves); V block staged
// TRANSPOSED (d-major) so P@V B-fragments are aligned ds_load_b128.
// All fragments for a WMMA chain are loaded into distinct registers first so
// the ds_load_b128s issue as one clause and overlap with the matrix ops.
// ---------------------------------------------------------------------------
__global__ __launch_bounds__(128) void attn_kernel(const __bf16* __restrict__ qkv,
                                                   __bf16* __restrict__ y) {
  __shared__ __align__(16) __bf16 sK[64 * 72];     // keys x hd (row-major)
  __shared__ __align__(16) __bf16 sVt[64 * 72];    // hd x keys (transposed)
  __shared__ __align__(16) __bf16 sP[4][16 * 72];  // per-wave P transpose buf

  const int lane = threadIdx.x & 31, w = threadIdx.x >> 5;
  const int lm = lane & 15, lh = lane >> 4;
  const int qb = blockIdx.x & 15;
  const int h  = (blockIdx.x >> 4) & 15;
  const int b  = blockIdx.x >> 8;
  const int q0 = qb * 64;
  const float scale = 0.125f;  // 1/sqrt(64)

  const __bf16* qbase = qkv + (size_t)(b * Tt) * S3 + h * HD;
  const __bf16* kbase = qbase + Ee;
  const __bf16* vbase = qbase + 2 * Ee;

  // Q fragments for this wave's 16 rows (hd = 64 -> 2 K-steps)
  const __bf16* qp = qbase + (size_t)(q0 + w * 16 + lm) * S3;
  const v16bf qf0 = ld_frag(qp, lh * 8);
  const v16bf qf1 = ld_frag(qp + 32, lh * 8);

  float mi[8], li[8];
  v8f o[4] = {};
#pragma unroll
  for (int r = 0; r < 8; ++r) { mi[r] = -INFINITY; li[r] = 0.0f; }

  const int nkb = q0 / 64 + 1;  // causal: key blocks 0 .. diagonal
  for (int kb = 0; kb < nkb; ++kb) {
    __syncthreads();  // previous iteration's sK/sVt reads complete
    {
      const int tr = threadIdx.x >> 1, tc = (threadIdx.x & 1) * 32;
      // K block: coalesced rows -> row-major LDS
      const __bf16* kp = kbase + (size_t)(kb * 64 + tr) * S3 + tc;
      __bf16* dk = &sK[tr * 72 + tc];
#pragma unroll
      for (int j = 0; j < 4; ++j) *(v8bf*)(dk + j * 8) = *(const v8bf*)(kp + j * 8);
      // V block: coalesced row read, transposed scalar scatter into LDS
      const __bf16* vp = vbase + (size_t)(kb * 64 + tr) * S3 + tc;
      v8bf v0 = *(const v8bf*)(vp);
      v8bf v1 = *(const v8bf*)(vp + 8);
      v8bf v2 = *(const v8bf*)(vp + 16);
      v8bf v3 = *(const v8bf*)(vp + 24);
#pragma unroll
      for (int j = 0; j < 8; ++j) {
        sVt[(tc + j) * 72 + tr]      = v0[j];
        sVt[(tc + 8 + j) * 72 + tr]  = v1[j];
        sVt[(tc + 16 + j) * 72 + tr] = v2[j];
        sVt[(tc + 24 + j) * 72 + tr] = v3[j];
      }
    }
    __syncthreads();

    // S = Q @ K^T : batch all 8 K-fragment loads, then 8 back-to-back WMMAs
    float s[4][8];
    {
      v16bf kf[4][2];
#pragma unroll
      for (int nt = 0; nt < 4; ++nt) {
        const __bf16* kr = &sK[(nt * 16 + lm) * 72];
        kf[nt][0] = ld_frag(kr, lh * 8);
        kf[nt][1] = ld_frag(kr + 32, lh * 8);
      }
#pragma unroll
      for (int nt = 0; nt < 4; ++nt) {
        v8f a = {};
        a = wmma_bf16(qf0, kf[nt][0], a);
        a = wmma_bf16(qf1, kf[nt][1], a);
#pragma unroll
        for (int r = 0; r < 8; ++r) {
          const int ng = kb * 64 + nt * 16 + lm;
          const int mg = q0 + w * 16 + r + 8 * lh;
          s[nt][r] = (ng > mg) ? -INFINITY : a[r] * scale;
        }
      }
    }

    // Online softmax. Row m lives across one 16-lane half -> shfl_xor width 16.
    float rm[8];
#pragma unroll
    for (int r = 0; r < 8; ++r) {
      float v = fmaxf(fmaxf(s[0][r], s[1][r]), fmaxf(s[2][r], s[3][r]));
      v = fmaxf(v, __shfl_xor(v, 1, 16));
      v = fmaxf(v, __shfl_xor(v, 2, 16));
      v = fmaxf(v, __shfl_xor(v, 4, 16));
      v = fmaxf(v, __shfl_xor(v, 8, 16));
      rm[r] = v;
    }
    float alpha[8], rs[8];
#pragma unroll
    for (int r = 0; r < 8; ++r) {
      const float mn = fmaxf(mi[r], rm[r]);
      alpha[r] = __expf(mi[r] - mn);
      mi[r] = mn;
      rs[r] = 0.0f;
    }
#pragma unroll
    for (int nt = 0; nt < 4; ++nt)
#pragma unroll
      for (int r = 0; r < 8; ++r) {
        const float p = __expf(s[nt][r] - mi[r]);
        s[nt][r] = p;
        rs[r] += p;
      }
#pragma unroll
    for (int r = 0; r < 8; ++r) {
      float v = rs[r];
      v += __shfl_xor(v, 1, 16);
      v += __shfl_xor(v, 2, 16);
      v += __shfl_xor(v, 4, 16);
      v += __shfl_xor(v, 8, 16);
      li[r] = li[r] * alpha[r] + v;
    }
#pragma unroll
    for (int dt = 0; dt < 4; ++dt)
#pragma unroll
      for (int r = 0; r < 8; ++r) o[dt][r] *= alpha[r];

    // P (C layout) -> LDS row-major, then reload as A fragments
#pragma unroll
    for (int nt = 0; nt < 4; ++nt)
#pragma unroll
      for (int r = 0; r < 8; ++r)
        sP[w][(r + 8 * lh) * 72 + nt * 16 + lm] = (__bf16)s[nt][r];
    __syncthreads();
    const __bf16* pp = &sP[w][lm * 72];
    const v16bf pf0 = ld_frag(pp, lh * 8);
    const v16bf pf1 = ld_frag(pp + 32, lh * 8);

    // O += P @ V : batch all 8 V-fragment loads, then 8 back-to-back WMMAs
    {
      v16bf vf[4][2];
#pragma unroll
      for (int dt = 0; dt < 4; ++dt) {
        const __bf16* vr = &sVt[(dt * 16 + lm) * 72];
        vf[dt][0] = ld_frag(vr, lh * 8);
        vf[dt][1] = ld_frag(vr + 32, lh * 8);
      }
#pragma unroll
      for (int dt = 0; dt < 4; ++dt) {
        o[dt] = wmma_bf16(pf0, vf[dt][0], o[dt]);
        o[dt] = wmma_bf16(pf1, vf[dt][1], o[dt]);
      }
    }
  }

  // normalize and emit bf16 for the projection GEMM
#pragma unroll
  for (int dt = 0; dt < 4; ++dt)
#pragma unroll
    for (int r = 0; r < 8; ++r) {
      const int t = q0 + w * 16 + r + 8 * lh;
      const int d = dt * 16 + lm;
      y[(size_t)(b * Tt + t) * Ee + h * HD + d] = (__bf16)(o[dt][r] / li[r]);
    }
}

// ---------------------------------------------------------------------------
// LayerNorm (f32 in) -> bf16 out. One block per row of E=1024.
// ---------------------------------------------------------------------------
__global__ __launch_bounds__(256) void ln_bf16_kernel(const float* __restrict__ x,
                                                      const float* __restrict__ w,
                                                      const float* __restrict__ bb,
                                                      __bf16* __restrict__ out) {
  __shared__ float s1[256], s2[256];
  const int row = blockIdx.x, tid = threadIdx.x;
  const float* xr = x + (size_t)row * Ee;
  float sum = 0.f, sq = 0.f;
  for (int j = tid; j < Ee; j += 256) {
    const float v = xr[j];
    sum += v;
    sq += v * v;
  }
  s1[tid] = sum;
  s2[tid] = sq;
  __syncthreads();
  for (int st = 128; st > 0; st >>= 1) {
    if (tid < st) { s1[tid] += s1[tid + st]; s2[tid] += s2[tid + st]; }
    __syncthreads();
  }
  const float mean = s1[0] * (1.0f / Ee);
  const float var  = s2[0] * (1.0f / Ee) - mean * mean;
  const float inv  = rsqrtf(var + 1e-5f);
  for (int j = tid; j < Ee; j += 256)
    out[(size_t)row * Ee + j] = (__bf16)((xr[j] - mean) * inv * w[j] + bb[j]);
}

// ---------------------------------------------------------------------------
// Embedding: x[b,t,:] = wte[idx[b,t],:] + wpe[t,:]
// ---------------------------------------------------------------------------
__global__ __launch_bounds__(256) void embed_kernel(const int* __restrict__ idx,
                                                    const float* __restrict__ wte,
                                                    const float* __restrict__ wpe,
                                                    float* __restrict__ x) {
  const int bt = blockIdx.x;
  const int t  = bt & (Tt - 1);
  const int tok = idx[bt];
  for (int j = threadIdx.x; j < Ee; j += 256)
    x[(size_t)bt * Ee + j] = wte[(size_t)tok * Ee + j] + wpe[(size_t)t * Ee + j];
}

// ---------------------------------------------------------------------------
extern "C" void kernel_launch(void* const* d_in, const int* in_sizes, int n_in,
                              void* d_out, int out_size, void* d_ws, size_t ws_size,
                              hipStream_t stream) {
  (void)in_sizes; (void)n_in; (void)out_size; (void)ws_size;
  const int*   idx      = (const int*)d_in[0];
  const float* wte      = (const float*)d_in[1];
  const float* wpe      = (const float*)d_in[2];
  const float* ln1_w    = (const float*)d_in[3];
  const float* ln1_b    = (const float*)d_in[4];
  const float* attn_w   = (const float*)d_in[5];
  const float* attn_b   = (const float*)d_in[6];
  const float* proj_w   = (const float*)d_in[7];
  const float* proj_b   = (const float*)d_in[8];
  const float* ln2_w    = (const float*)d_in[9];
  const float* ln2_b    = (const float*)d_in[10];
  const float* fc_w     = (const float*)d_in[11];
  const float* fc_b     = (const float*)d_in[12];
  const float* fc2_w    = (const float*)d_in[13];
  const float* fc2_b    = (const float*)d_in[14];
  const float* lnf_w    = (const float*)d_in[15];
  const float* lnf_b    = (const float*)d_in[16];
  const float* lm_head  = (const float*)d_in[17];

  char* ws = (char*)d_ws;
  float*  x      = (float*)ws;                                   // 16 MB
  __bf16* h_bf   = (__bf16*)(ws + (size_t)16 * 1024 * 1024);     //  8 MB
  __bf16* qkv_bf = (__bf16*)(ws + (size_t)24 * 1024 * 1024);     // 24 MB
  __bf16* y_bf   = (__bf16*)(ws + (size_t)48 * 1024 * 1024);     //  8 MB
  __bf16* u_bf   = (__bf16*)(ws + (size_t)56 * 1024 * 1024);     // 32 MB (total 88)
  float* logits  = (float*)d_out;

  embed_kernel<<<Mrows, 256, 0, stream>>>(idx, wte, wpe, x);

  for (int l = 0; l < Ll; ++l) {
    ln_bf16_kernel<<<Mrows, 256, 0, stream>>>(x, ln1_w + l * Ee, ln1_b + l * Ee, h_bf);
    gemm_bf16_kernel<0><<<dim3(S3 / 128, Mrows / 128), 256, 0, stream>>>(
        h_bf, attn_w + (size_t)l * S3 * Ee, attn_b + (size_t)l * S3, nullptr,
        qkv_bf, nullptr, S3, Ee);
    attn_kernel<<<Bz * Hh * (Tt / 64), 128, 0, stream>>>(qkv_bf, y_bf);
    gemm_bf16_kernel<2><<<dim3(Ee / 128, Mrows / 128), 256, 0, stream>>>(
        y_bf, proj_w + (size_t)l * Ee * Ee, proj_b + (size_t)l * Ee, x,
        nullptr, x, Ee, Ee);
    ln_bf16_kernel<<<Mrows, 256, 0, stream>>>(x, ln2_w + l * Ee, ln2_b + l * Ee, h_bf);
    gemm_bf16_kernel<1><<<dim3(4 * Ee / 128, Mrows / 128), 256, 0, stream>>>(
        h_bf, fc_w + (size_t)l * 4 * Ee * Ee, fc_b + (size_t)l * 4 * Ee, nullptr,
        u_bf, nullptr, 4 * Ee, Ee);
    gemm_bf16_kernel<2><<<dim3(Ee / 128, Mrows / 128), 256, 0, stream>>>(
        u_bf, fc2_w + (size_t)l * Ee * 4 * Ee, fc2_b + (size_t)l * Ee, x,
        nullptr, x, Ee, 4 * Ee);
  }

  ln_bf16_kernel<<<Mrows, 256, 0, stream>>>(x, lnf_w, lnf_b, h_bf);
  gemm_bf16_kernel<3><<<dim3(Vv / 128, Mrows / 128), 256, 0, stream>>>(
      h_bf, lm_head, nullptr, nullptr, nullptr, logits, Vv, Ee);
}
